// RPN_24378234372115
// MI455X (gfx1250) — matrix operand order repaired and checked by reading, not compiled, not run on hardware
//
#include <hip/hip_runtime.h>

// ---------------------------------------------------------------------------
// RPN forward for MI455X (gfx1250, wave32, WMMA).
//   conv3x3 as implicit GEMM: 9 x ([512x1024] x [1024x3750]), ~35 GFLOP,
//     compute-bound (AI ~ 1000 FLOP/B) -> f16 WMMA, f32 accumulate,
//     4x4 register blocking (16 wmma / K-step per wave), 32-bit offsets,
//     launch_bounds(256,1) for the full VGPR budget.
//   heads: [64x512] x [512x3750] WMMA GEMM, double-buffered LDS pipeline
//     fed by gfx1250 global_load_async_to_lds_b128 (ASYNCcnt in-order:
//     s_wait_asynccnt 2 retires the previous stage while the next flies).
// Everything else is scalar VALU work.
// ---------------------------------------------------------------------------

typedef __attribute__((ext_vector_type(16))) _Float16 v16h;
typedef __attribute__((ext_vector_type(8)))  _Float16 v8h;
typedef __attribute__((ext_vector_type(8)))  float    v8f;
typedef __attribute__((ext_vector_type(4)))  int      v4i;

#define HH 50
#define WW 75
#define PP 3750            // H*W spatial positions
#define CIN 1024
#define CMID 512
#define NA 33750           // PP * 9 anchors
#define NGT 20
#define PRE_NMS 6000
#define POST_NMS 300
#define SORTN 65536

// conv GEMM tiling: wave = 64(M) x 64(N), block = 8 waves
#define CONV_MB 8          // 512/64
#define CONV_NB 59         // ceil(3750/64)
// head GEMM tiling: block = 64(M) x 128(N), wave = 16(M) x 64(N)
#define HEAD_NB 30         // ceil(3750/128)

#if defined(__has_builtin)
#if __has_builtin(__builtin_amdgcn_global_load_async_to_lds_b128)
#define HAVE_ASYNC_LDS 1
#endif
#if __has_builtin(__builtin_amdgcn_s_wait_asynccnt)
#define HAVE_WAIT_ASYNC 1
#endif
#endif

#if defined(HAVE_ASYNC_LDS)
// 32 bytes per thread per stage = two b128 async copies (ASYNCcnt += 2).
#define ASYNC_CP32(g_, d_)                                                    \
  do {                                                                        \
    __builtin_amdgcn_global_load_async_to_lds_b128(                           \
        (__attribute__((address_space(1))) v4i*)(g_),                         \
        (__attribute__((address_space(3))) v4i*)(d_), 0, 0);                  \
    __builtin_amdgcn_global_load_async_to_lds_b128(                           \
        (__attribute__((address_space(1))) v4i*)((g_) + 8),                   \
        (__attribute__((address_space(3))) v4i*)((d_) + 8), 0, 0);            \
  } while (0)
#if defined(HAVE_WAIT_ASYNC)
#define ASYNC_WAIT(n) __builtin_amdgcn_s_wait_asynccnt(n)
#else
#define ASYNC_WAIT(n)
#endif
#else
#define ASYNC_CP32(g_, d_) (*(v16h*)(d_) = *(const v16h*)(g_))
#define ASYNC_WAIT(n)
#endif

// ----------------------------- helpers -------------------------------------

__device__ __forceinline__ unsigned fkey(float f) {
  unsigned u = __float_as_uint(f);
  return (u & 0x80000000u) ? ~u : (u | 0x80000000u);
}
__device__ __forceinline__ float funkey(unsigned k) {
  unsigned u = (k & 0x80000000u) ? (k ^ 0x80000000u) : ~k;
  return __uint_as_float(u);
}

__device__ __forceinline__ float iou1(float a0, float a1, float a2, float a3,
                                      const float* g) {
  float iw = fminf(a2, g[2]) - fmaxf(a0, g[0]) + 1.f;
  float ih = fminf(a3, g[3]) - fmaxf(a1, g[1]) + 1.f;
  if (iw <= 0.f || ih <= 0.f) return 0.f;
  float inter = iw * ih;
  float aa = (a2 - a0 + 1.f) * (a3 - a1 + 1.f);
  float ab = (g[2] - g[0] + 1.f) * (g[3] - g[1] + 1.f);
  return inter / (aa + ab - inter);
}

__device__ __forceinline__ void base_anchor(int k, float* a) {
  const float ratios[3] = {0.5f, 1.f, 2.f};
  const float scales[3] = {8.f, 16.f, 32.f};
  int ri = k / 3, si = k % 3;
  float xc = 7.5f, yc = 7.5f, size = 256.f;
  float ws0 = rintf(sqrtf(size / ratios[ri]));
  float hs0 = rintf(ws0 * ratios[ri]);
  float x1 = xc - 0.5f * (ws0 - 1.f), y1 = yc - 0.5f * (hs0 - 1.f);
  float x2 = xc + 0.5f * (ws0 - 1.f), y2 = yc + 0.5f * (hs0 - 1.f);
  float w = x2 - x1 + 1.f, h = y2 - y1 + 1.f;
  float cx = x1 + 0.5f * (w - 1.f), cy = y1 + 0.5f * (h - 1.f);
  float W2 = w * scales[si], H2 = h * scales[si];
  a[0] = cx - 0.5f * (W2 - 1.f);
  a[1] = cy - 0.5f * (H2 - 1.f);
  a[2] = cx + 0.5f * (W2 - 1.f);
  a[3] = cy + 0.5f * (H2 - 1.f);
}

__device__ __forceinline__ v16h frag_a(const _Float16* row, int hf) {
  // 16-bit A 16x32, per-lane contiguous K chunk at 16*hf (matches B).
  v8h lo = *(const v8h*)(row + 16 * hf);
  v8h hi = *(const v8h*)(row + 16 * hf + 8);
  return __builtin_shufflevector(lo, hi, 0, 1, 2, 3, 4, 5, 6, 7,
                                 8, 9, 10, 11, 12, 13, 14, 15);
}

// ------------------------- precision conversion ----------------------------

__global__ void k_cvt_feat(const float* __restrict__ src, _Float16* __restrict__ Ft) {
  int i = blockIdx.x * blockDim.x + threadIdx.x;
  if (i >= CIN * PP) return;
  int c = i / PP, p = i % PP;
  Ft[(size_t)p * CIN + c] = (_Float16)src[i];
}

// conv_w OIHW [512][1024][3][3] -> Wc [t][o][c] f16, t = ky*3+kx.
__global__ void k_cvt_wconv(const float* __restrict__ src, _Float16* __restrict__ Wc) {
  int i = blockIdx.x * blockDim.x + threadIdx.x;
  if (i >= CMID * CIN * 9) return;
  int o = i / (CIN * 9);
  int rem = i % (CIN * 9);
  int c = rem / 9, t = rem % 9;
  Wc[((size_t)t * CMID + o) * CIN + c] = (_Float16)src[i];
}

__global__ void k_cvt_whead(const float* __restrict__ cls_w,
                            const float* __restrict__ bbox_w,
                            _Float16* __restrict__ Wh) {
  int i = blockIdx.x * blockDim.x + threadIdx.x;
  if (i >= 64 * CMID) return;
  int m = i >> 9, c = i & 511;
  float v = (m < 18) ? cls_w[m * CMID + c]
                     : (m < 54 ? bbox_w[(m - 18) * CMID + c] : 0.f);
  Wh[i] = (_Float16)v;
}

__global__ void k_fill_hb(const float* __restrict__ cls_b,
                          const float* __restrict__ bbox_b,
                          float* __restrict__ hb) {
  int m = threadIdx.x;
  if (m >= 64) return;
  hb[m] = (m < 18) ? cls_b[m] : (m < 54 ? bbox_b[m - 18] : 0.f);
}

// ------------------------------ WMMA conv ----------------------------------
// Wave computes a 64x64 tile of [Cout x spatial] with 4x4 register blocking:
// 16 independent v_wmma_f32_16x16x32_f16 per K-step. B fragments loaded
// first; each A fragment is loaded and immediately consumed by its 4 wmma
// to keep peak VGPR liveness at 1A + 4B + 16 acc.
__global__ void __launch_bounds__(256, 1)
k_conv_wmma(const _Float16* __restrict__ Wc,   // [9][512][1024]
            const _Float16* __restrict__ Ft,   // [3750][1024]
            const float* __restrict__ bias,    // [512]
            _Float16* __restrict__ Xt) {       // [3750][512]
  int wave = threadIdx.x >> 5;
  int lane = threadIdx.x & 31;
  int tile = blockIdx.x * 8 + wave;         // 59 blocks * 8 waves = 472 exact
  int mb = tile % CONV_MB;
  int nb = tile / CONV_MB;
  int hf = lane >> 4;
  int l16 = lane & 15;

  // Per-lane A row byte offsets (output channels) for the 4 M sub-tiles.
  unsigned aoff[4];
#pragma unroll
  for (int im = 0; im < 4; ++im)
    aoff[im] = (unsigned)(mb * 64 + im * 16 + l16) * (CIN * 2);

  // Per-lane B columns (spatial) for the 4 N sub-tiles.
  int ncol[4], yy0[4], xx0[4];
#pragma unroll
  for (int in = 0; in < 4; ++in) {
    int n = nb * 64 + in * 16 + l16;
    ncol[in] = n;
    int np = n < PP ? n : PP - 1;
    yy0[in] = np / WW;
    xx0[in] = np % WW;
  }

  v8f acc[4][4] = {};

#pragma unroll 1
  for (int t = 0; t < 9; ++t) {
    int dy = t / 3 - 1, dx = t % 3 - 1;
    unsigned toff = (unsigned)t * (CMID * CIN * 2);
    unsigned boff[4];
    bool bv[4];
#pragma unroll
    for (int in = 0; in < 4; ++in) {
      int yy = yy0[in] + dy, xx = xx0[in] + dx;
      bv[in] = (yy >= 0) && (yy < HH) && (xx >= 0) && (xx < WW);
      boff[in] = (unsigned)(yy * WW + xx) * (CIN * 2);
    }
#pragma unroll 1
    for (int cb = 0; cb < CIN; cb += 32) {
      v16h bfr[4];
#pragma unroll
      for (int in = 0; in < 4; ++in) {
        v16h b = {};
        if (bv[in])
          b = *(const v16h*)((const char*)Ft + boff[in] + (cb + 16 * hf) * 2);
        bfr[in] = b;
      }
      __builtin_prefetch((const char*)Wc + (toff + aoff[0]) + cb * 2 + 256, 0, 0);
#pragma unroll
      for (int im = 0; im < 4; ++im) {
        const _Float16* ap =
            (const _Float16*)((const char*)Wc + (toff + aoff[im]) + cb * 2);
        v16h afr = frag_a(ap, hf);
#pragma unroll
        for (int in = 0; in < 4; ++in)
          acc[im][in] = __builtin_amdgcn_wmma_f32_16x16x32_f16(
              false, afr, false, bfr[in], (short)0, acc[im][in], false, false);
      }
    }
  }

#pragma unroll
  for (int in = 0; in < 4; ++in) {
    int n = ncol[in];
    if (n >= PP) continue;
#pragma unroll
    for (int im = 0; im < 4; ++im) {
#pragma unroll
      for (int r = 0; r < 8; ++r) {
        int mo = mb * 64 + im * 16 + r + 8 * hf;   // C/D: M = r + 8*half
        float v = acc[im][in][r] + bias[mo];
        v = v > 0.f ? v : 0.f;                     // ReLU
        Xt[(size_t)n * CMID + mo] = (_Float16)v;
      }
    }
  }
}

// ------------------------------ WMMA heads ---------------------------------
// Block computes [64 x 128] of Hout. Double-buffered B tile (K=32 x N=128,
// 8 KB per buffer) streamed global->LDS with async copies; each thread keeps
// 2 async b128 in flight per stage, so s_wait_asynccnt 2 retires stage s
// while stage s+1 is still flying. Each wave: 16(M) x 64(N).
__global__ void __launch_bounds__(256, 1)
k_head_wmma(const _Float16* __restrict__ Wh,  // [64][512]
            const _Float16* __restrict__ Xt,  // [3750][512]
            const float* __restrict__ hb,     // [64]
            float* __restrict__ Hout) {       // [64][3750]
  __shared__ _Float16 Bs[2][128 * 32];   // [buf][col][k]
  int tid = threadIdx.x;
  int wave = tid >> 5;
  int lane = tid & 31;
  int hf = lane >> 4, l16 = lane & 15;
  int nb = blockIdx.x;
  int mtile = wave & 3;               // 4 M tiles
  int nh = wave >> 2;                 // 2 N halves of 64

  // staging assignment: each thread moves 32 bytes per K-step
  int scol = tid >> 1;                // 0..127
  int spart = tid & 1;                // 0..1 (16-half chunks)
  int sn = nb * 128 + scol;
  int snp = sn < PP ? sn : PP - 1;
  const _Float16* sgp0 = Xt + (size_t)snp * CMID + 16 * spart;
  _Float16* sl0 = &Bs[0][scol * 32 + 16 * spart];
  _Float16* sl1 = &Bs[1][scol * 32 + 16 * spart];

  const _Float16* arow = Wh + (size_t)(mtile * 16 + l16) * CMID;

  const int NSTEP = CMID / 32;        // 16
  v8f acc[4] = {};

  ASYNC_CP32(sgp0, sl0);              // prime stage 0
#pragma unroll 1
  for (int s = 0; s < NSTEP; ++s) {
    if (s + 1 < NSTEP) {
      ASYNC_CP32(sgp0 + (s + 1) * 32, ((s + 1) & 1) ? sl1 : sl0);
      ASYNC_WAIT(2);                  // stage s done (in-order), s+1 in flight
    } else {
      ASYNC_WAIT(0);
    }
    __syncthreads();
    v16h a = frag_a(arow + s * 32, hf);
    const _Float16* bb = Bs[s & 1];
#pragma unroll
    for (int in = 0; in < 4; ++in) {
      v16h b = *(const v16h*)&bb[(nh * 64 + in * 16 + l16) * 32 + 16 * hf];
      acc[in] = __builtin_amdgcn_wmma_f32_16x16x32_f16(
          false, a, false, b, (short)0, acc[in], false, false);
    }
    __syncthreads();                  // all reads done before buf reuse
  }

#pragma unroll
  for (int in = 0; in < 4; ++in) {
    int n = nb * 128 + nh * 64 + in * 16 + l16;
    if (n >= PP) continue;
#pragma unroll
    for (int r = 0; r < 8; ++r) {
      int m = mtile * 16 + r + 8 * hf;
      Hout[(size_t)m * PP + n] = acc[in][r] + hb[m];
    }
  }
}

// --------------------------- proposal generation ---------------------------

__global__ void k_proposals(const float* __restrict__ Hout,
                            const float* __restrict__ info,
                            float* __restrict__ props,
                            float* __restrict__ scores,
                            float* __restrict__ anch) {
  int i = blockIdx.x * blockDim.x + threadIdx.x;
  if (i >= NA) return;
  int p = i / 9, k = i % 9;
  int y = p / WW, x = p % WW;
  float ba[4];
  base_anchor(k, ba);
  float a0 = ba[0] + 16.f * x, a1 = ba[1] + 16.f * y;
  float a2 = ba[2] + 16.f * x, a3 = ba[3] + 16.f * y;
  anch[i * 4 + 0] = a0; anch[i * 4 + 1] = a1;
  anch[i * 4 + 2] = a2; anch[i * 4 + 3] = a3;
  float s0 = Hout[(size_t)k * PP + p];
  float s1 = Hout[(size_t)(9 + k) * PP + p];
  float sc = 1.f / (1.f + expf(s0 - s1));
  float d0 = Hout[(size_t)(18 + k * 4 + 0) * PP + p];
  float d1 = Hout[(size_t)(18 + k * 4 + 1) * PP + p];
  float d2 = Hout[(size_t)(18 + k * 4 + 2) * PP + p];
  float d3 = Hout[(size_t)(18 + k * 4 + 3) * PP + p];
  float w = a2 - a0 + 1.f, h = a3 - a1 + 1.f;
  float cx = a0 + 0.5f * w, cy = a1 + 0.5f * h;
  float px = d0 * w + cx, py = d1 * h + cy;
  float pw = expf(d2) * w, ph = expf(d3) * h;
  float imh = info[0], imw = info[1], ims = info[2];
  float x1 = fminf(fmaxf(px - 0.5f * pw, 0.f), imw - 1.f);
  float y1 = fminf(fmaxf(py - 0.5f * ph, 0.f), imh - 1.f);
  float x2 = fminf(fmaxf(px + 0.5f * pw, 0.f), imw - 1.f);
  float y2 = fminf(fmaxf(py + 0.5f * ph, 0.f), imh - 1.f);
  props[i * 4 + 0] = x1; props[i * 4 + 1] = y1;
  props[i * 4 + 2] = x2; props[i * 4 + 3] = y2;
  float bw = x2 - x1 + 1.f, bh = y2 - y1 + 1.f;
  float mins = 16.f * ims;
  scores[i] = (bw >= mins && bh >= mins) ? sc : -__builtin_inff();
}

// ------------------------- top-k via bitonic sort --------------------------

__global__ void k_sort_init(const float* __restrict__ scores,
                            float* __restrict__ key, int* __restrict__ val) {
  int i = blockIdx.x * blockDim.x + threadIdx.x;
  if (i >= SORTN) return;
  key[i] = (i < NA) ? scores[i] : -__builtin_inff();
  val[i] = (i < NA) ? i : 0;
}

__global__ void k_bitonic(float* __restrict__ key, int* __restrict__ val,
                          int kk, int j) {
  unsigned i = blockIdx.x * blockDim.x + threadIdx.x;
  unsigned ixj = i ^ (unsigned)j;
  if (ixj <= i) return;
  float a = key[i], b = key[ixj];
  bool swap = ((i & (unsigned)kk) == 0u) ? (a < b) : (a > b);  // descending
  if (swap) {
    key[i] = b; key[ixj] = a;
    int t = val[i]; val[i] = val[ixj]; val[ixj] = t;
  }
}

__global__ void k_gather_cand(const float* __restrict__ key,
                              const int* __restrict__ val,
                              const float* __restrict__ props,
                              float* __restrict__ cand,
                              float* __restrict__ cscore) {
  int i = blockIdx.x * blockDim.x + threadIdx.x;
  if (i >= PRE_NMS) return;
  int o = val[i];
  cscore[i] = key[i];
#pragma unroll
  for (int j = 0; j < 4; ++j) cand[i * 4 + j] = props[o * 4 + j];
}

// --------------------------------- NMS -------------------------------------

__global__ void k_nms(const float* __restrict__ cand,
                      const float* __restrict__ cscore,
                      int* __restrict__ kept) {
  __shared__ unsigned mask[(PRE_NMS + 31) / 32];
  __shared__ int flag;
  for (int w = threadIdx.x; w < (PRE_NMS + 31) / 32; w += blockDim.x) mask[w] = 0u;
  __syncthreads();
  for (int i = 0; i < PRE_NMS; ++i) {
    if (threadIdx.x == 0) flag = 0;
    __syncthreads();
    float ax1 = cand[i * 4 + 0], ay1 = cand[i * 4 + 1];
    float ax2 = cand[i * 4 + 2], ay2 = cand[i * 4 + 3];
    float aarea = (ax2 - ax1 + 1.f) * (ay2 - ay1 + 1.f);
    bool supp = false;
    for (int j = threadIdx.x; j < i && !supp; j += blockDim.x) {
      if (mask[j >> 5] & (1u << (j & 31))) {
        float bx1 = cand[j * 4 + 0], by1 = cand[j * 4 + 1];
        float bx2 = cand[j * 4 + 2], by2 = cand[j * 4 + 3];
        float iw = fminf(ax2, bx2) - fmaxf(ax1, bx1) + 1.f;
        float ih = fminf(ay2, by2) - fmaxf(ay1, by1) + 1.f;
        if (iw > 0.f && ih > 0.f) {
          float inter = iw * ih;
          float barea = (bx2 - bx1 + 1.f) * (by2 - by1 + 1.f);
          if (inter / (aarea + barea - inter) > 0.7f) supp = true;
        }
      }
    }
    if (supp) atomicOr(&flag, 1);
    __syncthreads();
    if (threadIdx.x == 0) {
      if (cscore[i] > -1e37f && flag == 0) mask[i >> 5] |= (1u << (i & 31));
    }
    __syncthreads();
  }
  for (int i = threadIdx.x; i < PRE_NMS; i += blockDim.x)
    kept[i] = (mask[i >> 5] >> (i & 31)) & 1;
}

__global__ void k_rois(const float* __restrict__ cand, const int* __restrict__ kept,
                       float* __restrict__ out) {
  for (int i = threadIdx.x; i < POST_NMS * 5; i += blockDim.x) out[i] = 0.f;
  __syncthreads();
  if (threadIdx.x == 0) {
    int r = 0;
    for (int i = 0; i < PRE_NMS && r < POST_NMS; ++i) {
      if (kept[i]) {
        out[r * 5 + 0] = 0.f;
        out[r * 5 + 1] = cand[i * 4 + 0];
        out[r * 5 + 2] = cand[i * 4 + 1];
        out[r * 5 + 3] = cand[i * 4 + 2];
        out[r * 5 + 4] = cand[i * 4 + 3];
        ++r;
      }
    }
  }
}

// -------------------------- anchor targets + losses ------------------------

__global__ void k_zero_small(float* __restrict__ acc, unsigned* __restrict__ gmax) {
  int t = threadIdx.x;
  if (t < 2) acc[t] = 0.f;
  if (t < NGT) gmax[t] = fkey(-1.f);
}

__global__ void k_anchor_stats(const float* __restrict__ anch,
                               const float* __restrict__ gt,
                               const float* __restrict__ info,
                               float* __restrict__ moArr, int* __restrict__ amaxArr,
                               int* __restrict__ insideArr,
                               unsigned* __restrict__ gmax) {
  int i = blockIdx.x * blockDim.x + threadIdx.x;
  if (i >= NA) return;
  float a0 = anch[i * 4 + 0], a1 = anch[i * 4 + 1];
  float a2 = anch[i * 4 + 2], a3 = anch[i * 4 + 3];
  float imh = info[0], imw = info[1];
  bool inside = (a0 >= 0.f) && (a1 >= 0.f) && (a2 < imw) && (a3 < imh);
  float mo = -2.f; int am = 0;
  for (int g = 0; g < NGT; ++g) {
    float v = inside ? iou1(a0, a1, a2, a3, gt + g * 5) : -1.f;
    if (v > mo) { mo = v; am = g; }
    atomicMax(&gmax[g], fkey(v));
  }
  moArr[i] = mo; amaxArr[i] = am; insideArr[i] = inside ? 1 : 0;
}

__global__ void k_labels(const float* __restrict__ anch, const float* __restrict__ gt,
                         const float* __restrict__ moArr,
                         const int* __restrict__ insideArr,
                         const unsigned* __restrict__ gmax,
                         float* __restrict__ labels) {
  int i = blockIdx.x * blockDim.x + threadIdx.x;
  if (i >= NA) return;
  bool inside = insideArr[i] != 0;
  float a0 = anch[i * 4 + 0], a1 = anch[i * 4 + 1];
  float a2 = anch[i * 4 + 2], a3 = anch[i * 4 + 3];
  bool best = false;
  if (inside) {
    for (int g = 0; g < NGT; ++g) {
      float v = iou1(a0, a1, a2, a3, gt + g * 5);
      if (v == funkey(gmax[g])) best = true;
    }
  }
  float mo = moArr[i];
  float lab = -1.f;
  if (inside && mo < 0.3f) lab = 0.f;
  if (best || (inside && mo >= 0.7f)) lab = 1.f;
  labels[i] = lab;
}

__global__ void k_subsample(float* __restrict__ labels, int* __restrict__ stats) {
  int cfg = 0;
  for (int i = 0; i < NA; ++i)
    if (labels[i] == 1.f) { ++cfg; if (cfg > 128) labels[i] = -1.f; }
  int nfg = cfg < 128 ? cfg : 128;
  int cap = 256 - nfg;
  int cbg = 0;
  for (int i = 0; i < NA; ++i)
    if (labels[i] == 0.f) { ++cbg; if (cbg > cap) labels[i] = -1.f; }
  int nl = 0;
  for (int i = 0; i < NA; ++i) if (labels[i] >= 0.f) ++nl;
  stats[0] = nl > 0 ? nl : 1;
}

__global__ void k_losses(const float* __restrict__ labels,
                         const int* __restrict__ amaxArr,
                         const float* __restrict__ anch,
                         const float* __restrict__ gt,
                         const float* __restrict__ Hout,
                         const int* __restrict__ stats,
                         float* __restrict__ acc) {
  int i = blockIdx.x * blockDim.x + threadIdx.x;
  if (i >= NA) return;
  float lab = labels[i];
  if (lab < 0.f) return;
  int p = i / 9, k = i % 9;
  float s0 = Hout[(size_t)k * PP + p];
  float s1 = Hout[(size_t)(9 + k) * PP + p];
  float mx = fmaxf(s0, s1);
  float lse = mx + logf(expf(s0 - mx) + expf(s1 - mx));
  float lp = (lab > 0.5f ? s1 : s0) - lse;
  atomicAdd(&acc[0], -lp);
  if (lab > 0.5f) {
    const float* g = gt + amaxArr[i] * 5;
    float a0 = anch[i * 4 + 0], a1 = anch[i * 4 + 1];
    float a2 = anch[i * 4 + 2], a3 = anch[i * 4 + 3];
    float ew = a2 - a0 + 1.f, eh = a3 - a1 + 1.f;
    float ecx = a0 + 0.5f * ew, ecy = a1 + 0.5f * eh;
    float gw = g[2] - g[0] + 1.f, gh = g[3] - g[1] + 1.f;
    float gcx = g[0] + 0.5f * gw, gcy = g[1] + 0.5f * gh;
    float t[4] = {(gcx - ecx) / ew, (gcy - ecy) / eh, logf(gw / ew), logf(gh / eh)};
    float s = 0.f;
#pragma unroll
    for (int j = 0; j < 4; ++j) {
      float d = Hout[(size_t)(18 + k * 4 + j) * PP + p] - t[j];
      float ad = fabsf(d);
      s += (ad < 1.f / 9.f) ? d * d * 4.5f : (ad - 0.5f / 9.f);
    }
    atomicAdd(&acc[1], s / (float)stats[0]);
  }
}

__global__ void k_finalize(const float* __restrict__ acc, const int* __restrict__ stats,
                           float* __restrict__ out) {
  out[POST_NMS * 5 + 0] = acc[0] / (float)stats[0];
  out[POST_NMS * 5 + 1] = acc[1];
}

// ------------------------------- launcher ----------------------------------

extern "C" void kernel_launch(void* const* d_in, const int* in_sizes, int n_in,
                              void* d_out, int out_size, void* d_ws, size_t ws_size,
                              hipStream_t stream) {
  (void)in_sizes; (void)n_in; (void)out_size; (void)ws_size;
  const float* base_feat = (const float*)d_in[0];
  const float* im_info   = (const float*)d_in[1];
  const float* gt_boxes  = (const float*)d_in[2];
  const float* conv_w    = (const float*)d_in[3];
  const float* conv_b    = (const float*)d_in[4];
  const float* cls_w     = (const float*)d_in[5];
  const float* cls_b     = (const float*)d_in[6];
  const float* bbox_w    = (const float*)d_in[7];
  const float* bbox_b    = (const float*)d_in[8];
  float* out = (float*)d_out;

  char* w = (char*)d_ws;
  size_t off = 0;
  auto alloc = [&](size_t bytes) -> char* {
    char* p = w + off;
    off += (bytes + 255) & ~(size_t)255;
    return p;
  };
  _Float16* Ft  = (_Float16*)alloc((size_t)PP * CIN * 2);
  _Float16* Wc  = (_Float16*)alloc((size_t)9 * CMID * CIN * 2);
  _Float16* Whd = (_Float16*)alloc((size_t)64 * CMID * 2);
  float*    hb  = (float*)alloc(64 * 4);
  _Float16* Xt  = (_Float16*)alloc((size_t)PP * CMID * 2);
  float*    Hout = (float*)alloc((size_t)64 * PP * 4);
  float*    props = (float*)alloc((size_t)NA * 4 * 4);
  float*    anch  = (float*)alloc((size_t)NA * 4 * 4);
  float*    scores = (float*)alloc((size_t)NA * 4);
  float*    skey = (float*)alloc((size_t)SORTN * 4);
  int*      sval = (int*)alloc((size_t)SORTN * 4);
  float*    cand = (float*)alloc((size_t)PRE_NMS * 4 * 4);
  float*    cscore = (float*)alloc((size_t)PRE_NMS * 4);
  int*      kept = (int*)alloc((size_t)PRE_NMS * 4);
  float*    moArr = (float*)alloc((size_t)NA * 4);
  int*      amaxArr = (int*)alloc((size_t)NA * 4);
  int*      insideArr = (int*)alloc((size_t)NA * 4);
  float*    labels = (float*)alloc((size_t)NA * 4);
  unsigned* gmax = (unsigned*)alloc(NGT * 4);
  float*    acc = (float*)alloc(2 * 4);
  int*      stats = (int*)alloc(2 * 4);

  // precision conversion / layout
  k_cvt_feat<<<dim3((CIN * PP + 255) / 256), 256, 0, stream>>>(base_feat, Ft);
  k_cvt_wconv<<<dim3((CMID * CIN * 9 + 255) / 256), 256, 0, stream>>>(conv_w, Wc);
  k_cvt_whead<<<dim3((64 * CMID + 255) / 256), 256, 0, stream>>>(cls_w, bbox_w, Whd);
  k_fill_hb<<<dim3(1), 64, 0, stream>>>(cls_b, bbox_b, hb);

  // WMMA GEMMs
  k_conv_wmma<<<dim3(CONV_NB), 256, 0, stream>>>(Wc, Ft, conv_b, Xt);
  k_head_wmma<<<dim3(HEAD_NB), 256, 0, stream>>>(Whd, Xt, hb, Hout);

  // proposals
  k_proposals<<<dim3((NA + 255) / 256), 256, 0, stream>>>(Hout, im_info, props, scores, anch);

  // top-k via bitonic sort (descending), 65536 keys
  k_sort_init<<<dim3(SORTN / 256), 256, 0, stream>>>(scores, skey, sval);
  for (int kk = 2; kk <= SORTN; kk <<= 1)
    for (int j = kk >> 1; j > 0; j >>= 1)
      k_bitonic<<<dim3(SORTN / 256), 256, 0, stream>>>(skey, sval, kk, j);
  k_gather_cand<<<dim3((PRE_NMS + 255) / 256), 256, 0, stream>>>(skey, sval, props, cand, cscore);

  // NMS + ROI emit
  k_nms<<<dim3(1), 1024, 0, stream>>>(cand, cscore, kept);
  k_rois<<<dim3(1), 256, 0, stream>>>(cand, kept, out);

  // anchor targets + losses
  k_zero_small<<<dim3(1), 64, 0, stream>>>(acc, gmax);
  k_anchor_stats<<<dim3((NA + 255) / 256), 256, 0, stream>>>(anch, gt_boxes, im_info,
                                                             moArr, amaxArr, insideArr, gmax);
  k_labels<<<dim3((NA + 255) / 256), 256, 0, stream>>>(anch, gt_boxes, moArr, insideArr,
                                                       gmax, labels);
  k_subsample<<<dim3(1), 1, 0, stream>>>(labels, stats);
  k_losses<<<dim3((NA + 255) / 256), 256, 0, stream>>>(labels, amaxArr, anch, gt_boxes,
                                                       Hout, stats, acc);
  k_finalize<<<dim3(1), 1, 0, stream>>>(acc, stats, out);
}
